// Model_20650202759556
// MI455X (gfx1250) — matrix operand order repaired and checked by reading, not compiled, not run on hardware
//
#include <hip/hip_runtime.h>
#include <cstdint>
#include <cstddef>

typedef __bf16 bf16;
typedef __attribute__((ext_vector_type(16))) __bf16 v16bf;
typedef __attribute__((ext_vector_type(8)))  float  v8f;
typedef __attribute__((ext_vector_type(4)))  int    v4i;

#define TB     4096   // tokens = B*L
#define BATCH  4
#define SEQ    1024
#define DMODEL 512
#define NHEAD  8
#define HSZ    64
#define NLAYER 6
#define KPAD   224    // 200 padded to multiple of 32
#define DFF    2048

#if defined(__AMDGCN__) && __has_builtin(__builtin_amdgcn_global_load_async_to_lds_b128)
#define USE_ASYNC_COPY 1
#else
#define USE_ASYNC_COPY 0
#endif

__device__ __forceinline__ float gelu_exact(float x) {
    return 0.5f * x * (1.0f + erff(x * 0.7071067811865475f));
}

union Frag { v16bf v; uint4 u[2]; bf16 e[16]; };

__device__ __forceinline__ v8f wmma_bf16(v16bf a, v16bf b, v8f c) {
    return __builtin_amdgcn_wmma_f32_16x16x32_bf16(false, a, false, b, (short)0, c, false, false);
}

// 16-byte global -> LDS chunk copy: async (ASYNCcnt) when available, else via VGPRs.
__device__ __forceinline__ void stage_chunk(const bf16* gsrc, bf16* ldst) {
#if USE_ASYNC_COPY
    __builtin_amdgcn_global_load_async_to_lds_b128(
        (v4i __attribute__((address_space(1)))*)const_cast<bf16*>(gsrc),
        (v4i __attribute__((address_space(3)))*)ldst,
        0, 0);
#else
    *(uint4*)ldst = *(const uint4*)gsrc;
#endif
}

__device__ __forceinline__ void wait_async_all() {
#if USE_ASYNC_COPY
#if __has_builtin(__builtin_amdgcn_s_wait_asynccnt)
    __builtin_amdgcn_s_wait_asynccnt(0);
#else
    asm volatile("s_wait_asynccnt 0x0" ::: "memory");
#endif
#endif
}

// ---------------- weight prep ----------------

__global__ __launch_bounds__(256)
void prep_wd_kernel(const float* __restrict__ Wd, bf16* __restrict__ out) {
    int idx = blockIdx.x * 256 + threadIdx.x;      // over 512*KPAD, out is [N=512][K=224]
    if (idx >= 512 * KPAD) return;
    int n = idx / KPAD, kk = idx % KPAD;
    out[idx] = (kk < 200) ? (bf16)Wd[(size_t)kk * 512 + n] : (bf16)0.0f;
}

__global__ __launch_bounds__(256)
void conv_t_kernel(const float* __restrict__ in, bf16* __restrict__ out, int K, int N) {
    size_t base = (size_t)blockIdx.y * K * N;      // [K][N] f32 -> [N][K] bf16 per slice
    int idx = blockIdx.x * 256 + threadIdx.x;
    if (idx >= K * N) return;
    int kk = idx / N, n = idx % N;
    out[base + (size_t)n * K + kk] = (bf16)in[base + idx];
}

__global__ __launch_bounds__(256)
void conv_p_kernel(const float* __restrict__ in, bf16* __restrict__ out, int n) {
    size_t base = (size_t)blockIdx.y * n;
    int idx = blockIdx.x * 256 + threadIdx.x;
    if (idx >= n) return;
    out[base + idx] = (bf16)in[base + idx];
}

// ---------------- embedding concat (padded to KPAD) ----------------

__global__ __launch_bounds__(256)
void embed_kernel(const int* __restrict__ stg, const int* __restrict__ egoc,
                  const int* __restrict__ oppc, const int* __restrict__ egoa,
                  const int* __restrict__ oppa, const float* __restrict__ gs,
                  const float* __restrict__ ctrl, const float* __restrict__ es,
                  const float* __restrict__ ec, const float* __restrict__ ea,
                  bf16* __restrict__ xpad) {
    int idx = blockIdx.x * 256 + threadIdx.x;
    if (idx >= TB * KPAD) return;
    int t = idx / KPAD, c = idx % KPAD;
    float v = 0.0f;
    if      (c < 8)   v = es[stg[t]  * 8  + c];
    else if (c < 24)  v = ec[egoc[t] * 16 + (c - 8)];
    else if (c < 40)  v = ec[oppc[t] * 16 + (c - 24)];
    else if (c < 72)  v = ea[egoa[t] * 32 + (c - 40)];
    else if (c < 104) v = ea[oppa[t] * 32 + (c - 72)];
    else if (c < 168) v = gs[(size_t)t * 64 + (c - 104)];
    else if (c < 200) v = ctrl[(size_t)t * 32 + (c - 168)];
    xpad[idx] = (bf16)v;
}

// ---------------- generic bf16 WMMA GEMM ----------------
// C[M,N](f32 accum) = A[M,K](bf16,row-major) * Bt[N,K]^T + bias, opt residual/GELU,
// writes f32 and/or bf16. Block: 128 thr (4 waves), tile 64x64, K-step 32,
// double-buffered async global->LDS staging of the B tile.

__global__ __launch_bounds__(128)
void gemm_bf16_kernel(const bf16* __restrict__ A, const bf16* __restrict__ Bt,
                      const float* __restrict__ bias, const float* __restrict__ res,
                      float* __restrict__ outF, bf16* __restrict__ outB,
                      int M, int N, int K, int act) {
    __shared__ __align__(16) bf16 ldsB[2][64 * 32];
    const int tid  = threadIdx.x;
    const int wave = tid >> 5;
    const int lane = tid & 31;
    const int lrow = lane & 15;
    const int hi   = lane >> 4;                 // 0 or 1
    const int n0      = blockIdx.x * 64;
    const int rowBase = blockIdx.y * 64 + wave * 16;
    v8f acc[4] = {};
    const bf16* Abase = A + (size_t)(rowBase + lrow) * K;

    // each thread stages 2 chunks of 8 bf16 per tile
    const int sr = tid >> 1;                    // 0..63 (tile row)
    const int sc = (tid & 1) * 16;              // 0 or 16 (tile col base)

    // prologue: stage tile k0=0 into buffer 0
    {
        const bf16* src = &Bt[(size_t)(n0 + sr) * K + sc];
        stage_chunk(src,     &ldsB[0][sr * 32 + sc]);
        stage_chunk(src + 8, &ldsB[0][sr * 32 + sc + 8]);
    }
    int cur = 0;
    for (int k0 = 0; k0 < K; k0 += 32) {
        wait_async_all();
        __syncthreads();
        if (k0 + 32 < K) {
            const bf16* src = &Bt[(size_t)(n0 + sr) * K + (k0 + 32) + sc];
            stage_chunk(src,     &ldsB[cur ^ 1][sr * 32 + sc]);
            stage_chunk(src + 8, &ldsB[cur ^ 1][sr * 32 + sc + 8]);
        }
        // A fragment (16x32): lane<16 -> k {0..15}, lane>=16 -> {16..31}
        Frag af;
        const bf16* Ap = Abase + k0 + hi * 16;
        af.u[0] = *(const uint4*)(Ap);
        af.u[1] = *(const uint4*)(Ap + 8);
        if (k0 + 32 < K) __builtin_prefetch(Abase + k0 + 32, 0, 1);
        #pragma unroll
        for (int nc = 0; nc < 4; ++nc) {
            Frag bfm;   // B fragment: col = lrow, k = hi*16 + e (contiguous in ldsB row)
            const bf16* Bp = &ldsB[cur][(nc * 16 + lrow) * 32 + hi * 16];
            bfm.u[0] = *(const uint4*)(Bp);
            bfm.u[1] = *(const uint4*)(Bp + 8);
            acc[nc] = wmma_bf16(af.v, bfm.v, acc[nc]);
        }
        cur ^= 1;
    }
    // epilogue: C element (M=r+hi*8, N=nc*16+lrow)
    #pragma unroll
    for (int nc = 0; nc < 4; ++nc) {
        int col = n0 + nc * 16 + lrow;
        float bv = bias ? bias[col] : 0.0f;
        #pragma unroll
        for (int r = 0; r < 8; ++r) {
            int row = rowBase + r + hi * 8;
            float v = acc[nc][r] + bv;
            if (res) v += res[(size_t)row * N + col];
            if (act == 1) v = gelu_exact(v);
            if (outF) outF[(size_t)row * N + col] = v;
            if (outB) outB[(size_t)row * N + col] = (bf16)v;
        }
    }
}

// ---------------- layernorm ----------------

__global__ __launch_bounds__(256)
void ln_kernel(const float* __restrict__ x, const float* __restrict__ w,
               const float* __restrict__ b, bf16* __restrict__ outB,
               float* __restrict__ outF, int D) {
    __shared__ float red[256];
    const int row = blockIdx.x, tid = threadIdx.x;
    const float* xr = x + (size_t)row * D;
    float s = 0.0f;
    for (int i = tid; i < D; i += 256) s += xr[i];
    red[tid] = s; __syncthreads();
    for (int st = 128; st > 0; st >>= 1) { if (tid < st) red[tid] += red[tid + st]; __syncthreads(); }
    float mu = red[0] / D;
    __syncthreads();
    float v = 0.0f;
    for (int i = tid; i < D; i += 256) { float d = xr[i] - mu; v += d * d; }
    red[tid] = v; __syncthreads();
    for (int st = 128; st > 0; st >>= 1) { if (tid < st) red[tid] += red[tid + st]; __syncthreads(); }
    float rstd = rsqrtf(red[0] / D + 1e-5f);
    __syncthreads();
    for (int i = tid; i < D; i += 256) {
        float o = (xr[i] - mu) * rstd * w[i] + b[i];
        if (outB) outB[(size_t)row * D + i] = (bf16)o;
        if (outF) outF[(size_t)row * D + i] = o;
    }
}

// ---------------- qkv split / v-transpose ----------------

__global__ __launch_bounds__(256)
void split_kernel(const bf16* __restrict__ qkvb, bf16* __restrict__ q,
                  bf16* __restrict__ k, bf16* __restrict__ vT) {
    int idx = blockIdx.x * 256 + threadIdx.x;
    if (idx >= TB * DMODEL) return;
    int t = idx >> 9, d = idx & 511;
    int h = d >> 6, hs = d & 63;
    int b = t >> 10, l = t & 1023;
    size_t src = (size_t)t * 1536;
    size_t bh  = (size_t)(b * NHEAD + h);
    size_t qdst = (bh * SEQ + l) * HSZ + hs;
    q[qdst] = qkvb[src + d];
    k[qdst] = qkvb[src + 512 + d];
    vT[(bh * HSZ + hs) * SEQ + l] = qkvb[src + 1024 + d];
}

// ---------------- fused causal attention with skewed rel-pos ----------------
// one wave per block; block = (b, h, 16-row query tile). LDS: scores[16][1024] f32.

__global__ __launch_bounds__(32)
void attn_kernel(const bf16* __restrict__ q, const bf16* __restrict__ k,
                 const bf16* __restrict__ vT, const bf16* __restrict__ ErB,
                 bf16* __restrict__ y) {
    extern __shared__ float smem[];
    float* s  = smem;              // 16*1024
    float* er = smem + 16 * 1024;  // 16*32
    const int lane = threadIdx.x;
    const int lrow = lane & 15;
    const int hi   = lane >> 4;
    const int lt = blockIdx.x & 63;
    const int h  = (blockIdx.x >> 6) & 7;
    const int b  = blockIdx.x >> 9;
    const int l0 = lt * 16;
    const size_t bh = (size_t)(b * NHEAD + h);
    const bf16* qp = q  + (bh * SEQ + l0) * HSZ;
    const bf16* kp = k  +  bh * SEQ * HSZ;
    const bf16* vp = vT +  bh * HSZ * SEQ;

    // Q fragments for both 32-wide hs chunks
    Frag qf[2];
    #pragma unroll
    for (int kc = 0; kc < 2; ++kc) {
        const bf16* p = qp + (size_t)lrow * HSZ + kc * 32 + hi * 16;
        qf[kc].u[0] = *(const uint4*)(p);
        qf[kc].u[1] = *(const uint4*)(p + 8);
    }
    const int ncols = l0 + 16;

    for (int j0 = 0; j0 < ncols; j0 += 16) {
        v8f acc = {};
        #pragma unroll
        for (int kc = 0; kc < 2; ++kc) {
            Frag bfm;   // keys: Bt row = key j0+lrow, contiguous hs
            const bf16* p = kp + (size_t)(j0 + lrow) * HSZ + kc * 32 + hi * 16;
            bfm.u[0] = *(const uint4*)(p);
            bfm.u[1] = *(const uint4*)(p + 8);
            acc = wmma_bf16(qf[kc].v, bfm.v, acc);
        }
        // rel-pos: S_rel[l,j] = q_l . Er[j-l+1023]; tile needs m in [m_base, m_base+31]
        int m_base = j0 - l0 + 1008;
        v8f eacc[2] = {};
        #pragma unroll
        for (int half = 0; half < 2; ++half) {
            int mrow = m_base + half * 16 + lrow;
            mrow = mrow < 0 ? 0 : (mrow > 1023 ? 1023 : mrow);
            #pragma unroll
            for (int kc = 0; kc < 2; ++kc) {
                Frag ef;
                const bf16* p = ErB + (size_t)mrow * HSZ + kc * 32 + hi * 16;
                ef.u[0] = *(const uint4*)(p);
                ef.u[1] = *(const uint4*)(p + 8);
                eacc[half] = wmma_bf16(qf[kc].v, ef.v, eacc[half]);
            }
        }
        #pragma unroll
        for (int r = 0; r < 8; ++r) {
            int row = r + hi * 8;
            s[row * 1024 + j0 + lrow] = acc[r];
            er[row * 32 + lrow]       = eacc[0][r];
            er[row * 32 + 16 + lrow]  = eacc[1][r];
        }
        __syncthreads();
        // combine skewed rel term + scale + causal mask (needs cross-lane -> via LDS)
        for (int idx = lane; idx < 256; idx += 32) {
            int r = idx >> 4, jj = idx & 15;
            int j = j0 + jj;
            int dm = jj - r + 15;               // in [0,30]
            float v = (s[r * 1024 + j] + er[r * 32 + dm]) * 0.125f;
            s[r * 1024 + j] = (j <= l0 + r) ? v : -3.0e38f;
        }
        __syncthreads();
    }

    // softmax per row (fp32)
    if (lane < 16) {
        int r = lane;
        float mx = -3.0e38f;
        for (int j = 0; j < ncols; ++j) mx = fmaxf(mx, s[r * 1024 + j]);
        float sum = 0.0f;
        for (int j = 0; j < ncols; ++j) {
            float e = __expf(s[r * 1024 + j] - mx);
            s[r * 1024 + j] = e; sum += e;
        }
        float inv = 1.0f / sum;
        for (int j = 0; j < ncols; ++j) s[r * 1024 + j] *= inv;
    }
    // zero-pad probs to a multiple of 32 keys
    if (ncols & 31) {
        for (int idx = lane; idx < 256; idx += 32) {
            int r = idx >> 4, jj = idx & 15;
            s[r * 1024 + ncols + jj] = 0.0f;
        }
    }
    __syncthreads();

    // O = P @ V  (vT rows = hs, contiguous over keys)
    int nround = (ncols + 31) & ~31;
    v8f o[4] = {};
    for (int j0 = 0; j0 < nround; j0 += 32) {
        Frag pf;
        const float* sp = s + lrow * 1024 + j0 + hi * 16;
        #pragma unroll
        for (int e = 0; e < 8; ++e) { pf.e[e] = (bf16)sp[e]; pf.e[8 + e] = (bf16)sp[8 + e]; }
        #pragma unroll
        for (int nc = 0; nc < 4; ++nc) {
            Frag bfm;
            const bf16* p = vp + (size_t)(nc * 16 + lrow) * SEQ + j0 + hi * 16;
            bfm.u[0] = *(const uint4*)(p);
            bfm.u[1] = *(const uint4*)(p + 8);
            o[nc] = wmma_bf16(pf.v, bfm.v, o[nc]);
        }
    }
    #pragma unroll
    for (int nc = 0; nc < 4; ++nc)
        #pragma unroll
        for (int r = 0; r < 8; ++r) {
            int row = l0 + r + hi * 8;
            y[((size_t)b * SEQ + row) * DMODEL + h * HSZ + nc * 16 + lrow] = (bf16)o[nc][r];
        }
}

// ---------------- cascaded output heads (small, VALU) ----------------

__global__ __launch_bounds__(256)
void head_kernel(const float* __restrict__ xf,
                 const float* __restrict__ p0, int w0,
                 const float* __restrict__ p1, int w1,
                 const float* __restrict__ p2, int w2,
                 const float* __restrict__ lnw, const float* __restrict__ lnb,
                 const float* __restrict__ W1, const float* __restrict__ b1,
                 const float* __restrict__ W2, const float* __restrict__ b2,
                 int d_in, int d_h, int d_out,
                 float* __restrict__ stage_out, float* __restrict__ fin, int out_off) {
    __shared__ float xin[576];
    __shared__ float hmid[288];
    __shared__ float red[256];
    const int t = blockIdx.x, tid = threadIdx.x;
    for (int i = tid; i < 512; i += 256) xin[i] = xf[(size_t)t * 512 + i];
    int off = 512;
    if (p0) { for (int i = tid; i < w0; i += 256) xin[off + i] = p0[(size_t)t * w0 + i]; off += w0; }
    if (p1) { for (int i = tid; i < w1; i += 256) xin[off + i] = p1[(size_t)t * w1 + i]; off += w1; }
    if (p2) { for (int i = tid; i < w2; i += 256) xin[off + i] = p2[(size_t)t * w2 + i]; off += w2; }
    __syncthreads();
    float s = 0.0f;
    for (int i = tid; i < d_in; i += 256) s += xin[i];
    red[tid] = s; __syncthreads();
    for (int st = 128; st > 0; st >>= 1) { if (tid < st) red[tid] += red[tid + st]; __syncthreads(); }
    float mu = red[0] / d_in;
    __syncthreads();
    float v = 0.0f;
    for (int i = tid; i < d_in; i += 256) { float d = xin[i] - mu; v += d * d; }
    red[tid] = v; __syncthreads();
    for (int st = 128; st > 0; st >>= 1) { if (tid < st) red[tid] += red[tid + st]; __syncthreads(); }
    float rstd = rsqrtf(red[0] / d_in + 1e-5f);
    __syncthreads();
    for (int i = tid; i < d_in; i += 256) xin[i] = (xin[i] - mu) * rstd * lnw[i] + lnb[i];
    __syncthreads();
    for (int hj = tid; hj < d_h; hj += 256) {
        float a = b1[hj];
        for (int i = 0; i < d_in; ++i) a += xin[i] * W1[(size_t)i * d_h + hj];
        hmid[hj] = gelu_exact(a);
    }
    __syncthreads();
    for (int oj = tid; oj < d_out; oj += 256) {
        float a = b2[oj];
        for (int i = 0; i < d_h; ++i) a += hmid[i] * W2[(size_t)i * d_out + oj];
        fin[(size_t)t * 58 + out_off + oj] = a;
        if (stage_out) stage_out[(size_t)t * d_out + oj] = a;
    }
}

// ---------------- host orchestration ----------------

extern "C" void kernel_launch(void* const* d_in, const int* in_sizes, int n_in,
                              void* d_out, int out_size, void* d_ws, size_t ws_size,
                              hipStream_t stream) {
    (void)in_sizes; (void)n_in; (void)out_size; (void)ws_size;
    const int*   stg  = (const int*)d_in[0];
    const int*   egoc = (const int*)d_in[1];
    const int*   oppc = (const int*)d_in[2];
    const int*   egoa = (const int*)d_in[3];
    const int*   oppa = (const int*)d_in[4];
    const float* gs   = (const float*)d_in[5];
    const float* ctrl = (const float*)d_in[6];
    const float* es   = (const float*)d_in[7];
    const float* ec   = (const float*)d_in[8];
    const float* ea   = (const float*)d_in[9];
    const float* Wd   = (const float*)d_in[10];
    const float* bd   = (const float*)d_in[11];
    const float* ln1w = (const float*)d_in[12];
    const float* ln1b = (const float*)d_in[13];
    const float* Wqkv = (const float*)d_in[14];
    const float* bqkv = (const float*)d_in[15];
    const float* Wo   = (const float*)d_in[16];
    const float* bo   = (const float*)d_in[17];
    const float* Er   = (const float*)d_in[18];
    const float* ln2w = (const float*)d_in[19];
    const float* ln2b = (const float*)d_in[20];
    const float* Wfc  = (const float*)d_in[21];
    const float* bfc  = (const float*)d_in[22];
    const float* Wp   = (const float*)d_in[23];
    const float* bp   = (const float*)d_in[24];
    const float* lnfw = (const float*)d_in[25];
    const float* lnfb = (const float*)d_in[26];
    const float* hp[4][6];
    for (int hh = 0; hh < 4; ++hh)
        for (int j = 0; j < 6; ++j)
            hp[hh][j] = (const float*)d_in[27 + hh * 6 + j];
    float* out = (float*)d_out;

    uint8_t* wsb = (uint8_t*)d_ws;
    size_t cur = 0;
    auto alloc = [&](size_t bytes) -> void* {
        cur = (cur + 255) & ~(size_t)255;
        void* p = wsb + cur;
        cur += bytes;
        return p;
    };
    bf16*  wd_t   = (bf16*)alloc((size_t)512 * KPAD * 2);
    bf16*  wqkv_t = (bf16*)alloc((size_t)NLAYER * 1536 * 512 * 2);
    bf16*  wo_t   = (bf16*)alloc((size_t)NLAYER * 512 * 512 * 2);
    bf16*  wfc_t  = (bf16*)alloc((size_t)NLAYER * 2048 * 512 * 2);
    bf16*  wp_t   = (bf16*)alloc((size_t)NLAYER * 512 * 2048 * 2);
    bf16*  er_b   = (bf16*)alloc((size_t)NLAYER * SEQ * HSZ * 2);
    bf16*  xpad   = (bf16*)alloc((size_t)TB * KPAD * 2);
    float* x      = (float*)alloc((size_t)TB * DMODEL * 4);
    bf16*  hbuf   = (bf16*)alloc((size_t)TB * DMODEL * 2);
    bf16*  qkvb   = (bf16*)alloc((size_t)TB * 1536 * 2);
    bf16*  qB     = (bf16*)alloc((size_t)TB * DMODEL * 2);
    bf16*  kB     = (bf16*)alloc((size_t)TB * DMODEL * 2);
    bf16*  vTb    = (bf16*)alloc((size_t)TB * DMODEL * 2);
    bf16*  yB     = (bf16*)alloc((size_t)TB * DMODEL * 2);
    bf16*  tB     = (bf16*)alloc((size_t)TB * DFF * 2);
    float* xf     = (float*)alloc((size_t)TB * DMODEL * 4);
    float* cbuf   = (float*)alloc((size_t)TB * 21 * 4);
    float* mbuf   = (float*)alloc((size_t)TB * 21 * 4);
    float* btbuf  = (float*)alloc((size_t)TB * 12 * 4);

    // --- weight conversion (f32 -> transposed bf16) ---
    prep_wd_kernel<<<(512 * KPAD + 255) / 256, 256, 0, stream>>>(Wd, wd_t);
    conv_t_kernel<<<dim3((512 * 1536 + 255) / 256, NLAYER), 256, 0, stream>>>(Wqkv, wqkv_t, 512, 1536);
    conv_t_kernel<<<dim3((512 * 512  + 255) / 256, NLAYER), 256, 0, stream>>>(Wo,   wo_t,   512, 512);
    conv_t_kernel<<<dim3((512 * 2048 + 255) / 256, NLAYER), 256, 0, stream>>>(Wfc,  wfc_t,  512, 2048);
    conv_t_kernel<<<dim3((2048 * 512 + 255) / 256, NLAYER), 256, 0, stream>>>(Wp,   wp_t,   2048, 512);
    conv_p_kernel<<<dim3((SEQ * HSZ + 255) / 256, NLAYER), 256, 0, stream>>>(Er, er_b, SEQ * HSZ);

    // --- embedding + input projection ---
    embed_kernel<<<(TB * KPAD + 255) / 256, 256, 0, stream>>>(stg, egoc, oppc, egoa, oppa,
                                                              gs, ctrl, es, ec, ea, xpad);
    gemm_bf16_kernel<<<dim3(512 / 64, TB / 64), 128, 0, stream>>>(
        xpad, wd_t, bd, nullptr, x, nullptr, TB, 512, KPAD, 0);

    // --- transformer layers ---
    const size_t attn_shmem = (size_t)(16 * 1024 + 16 * 32) * sizeof(float);
    for (int i = 0; i < NLAYER; ++i) {
        ln_kernel<<<TB, 256, 0, stream>>>(x, ln1w + i * 512, ln1b + i * 512, hbuf, nullptr, 512);
        gemm_bf16_kernel<<<dim3(1536 / 64, TB / 64), 128, 0, stream>>>(
            hbuf, wqkv_t + (size_t)i * 1536 * 512, bqkv + i * 1536,
            nullptr, nullptr, qkvb, TB, 1536, 512, 0);
        split_kernel<<<(TB * DMODEL + 255) / 256, 256, 0, stream>>>(qkvb, qB, kB, vTb);
        attn_kernel<<<BATCH * NHEAD * (SEQ / 16), 32, attn_shmem, stream>>>(
            qB, kB, vTb, er_b + (size_t)i * SEQ * HSZ, yB);
        gemm_bf16_kernel<<<dim3(512 / 64, TB / 64), 128, 0, stream>>>(
            yB, wo_t + (size_t)i * 512 * 512, bo + i * 512, x, x, nullptr, TB, 512, 512, 0);
        ln_kernel<<<TB, 256, 0, stream>>>(x, ln2w + i * 512, ln2b + i * 512, hbuf, nullptr, 512);
        gemm_bf16_kernel<<<dim3(2048 / 64, TB / 64), 128, 0, stream>>>(
            hbuf, wfc_t + (size_t)i * 2048 * 512, bfc + i * 2048,
            nullptr, nullptr, tB, TB, 2048, 512, 1);
        gemm_bf16_kernel<<<dim3(512 / 64, TB / 64), 128, 0, stream>>>(
            tB, wp_t + (size_t)i * 512 * 2048, bp + i * 512, x, x, nullptr, TB, 512, 2048, 0);
    }

    // --- final LN + cascaded heads ---
    ln_kernel<<<TB, 256, 0, stream>>>(x, lnfw, lnfb, nullptr, xf, 512);
    head_kernel<<<TB, 256, 0, stream>>>(xf, nullptr, 0, nullptr, 0, nullptr, 0,
        hp[0][0], hp[0][1], hp[0][2], hp[0][3], hp[0][4], hp[0][5],
        512, 256, 21, cbuf, out, 0);
    head_kernel<<<TB, 256, 0, stream>>>(xf, cbuf, 21, nullptr, 0, nullptr, 0,
        hp[1][0], hp[1][1], hp[1][2], hp[1][3], hp[1][4], hp[1][5],
        533, 266, 21, mbuf, out, 21);
    head_kernel<<<TB, 256, 0, stream>>>(xf, cbuf, 21, mbuf, 21, nullptr, 0,
        hp[2][0], hp[2][1], hp[2][2], hp[2][3], hp[2][4], hp[2][5],
        554, 277, 12, btbuf, out, 42);
    head_kernel<<<TB, 256, 0, stream>>>(xf, cbuf, 21, mbuf, 21, btbuf, 12,
        hp[3][0], hp[3][1], hp[3][2], hp[3][3], hp[3][4], hp[3][5],
        566, 283, 4, nullptr, out, 54);
}
